// DeformableDETR_71665824301091
// MI455X (gfx1250) — matrix operand order repaired and checked by reading, not compiled, not run on hardware
//
#include <hip/hip_runtime.h>
#include <hip/hip_bf16.h>

// Problem constants (from reference)
#define NBS 8
#define NL  4096
#define ND  768          // hidden dim D
#define NO  256          // expert output dim O
#define NE  8            // num experts
#define NTOK (NBS*NL)    // 32768 tokens
#define MT  64           // tokens per GEMM block tile

typedef __attribute__((ext_vector_type(16))) __bf16          v16bf;
typedef __attribute__((ext_vector_type(16))) unsigned short  v16u;
typedef __attribute__((ext_vector_type(8)))  unsigned short  v8u;
typedef __attribute__((ext_vector_type(8)))  float           v8f;

__device__ __forceinline__ unsigned short f2bf(float f) {
  // round-to-nearest-even fp32 -> bf16
  unsigned int u = __float_as_uint(f);
  u += 0x7fffu + ((u >> 16) & 1u);
  return (unsigned short)(u >> 16);
}

// ---------------------------------------------------------------------------
// Kernel 1: zero output accumulator + per-expert counters
// ---------------------------------------------------------------------------
__global__ __launch_bounds__(256) void moe_init(float* __restrict__ out,
                                                int* __restrict__ counts) {
  size_t i = (size_t)blockIdx.x * blockDim.x + threadIdx.x;
  if (i < (size_t)NTOK * NO / 4) {
    float4 z; z.x = 0.f; z.y = 0.f; z.z = 0.f; z.w = 0.f;
    reinterpret_cast<float4*>(out)[i] = z;
  }
  if (i < NE) counts[i] = 0;
}

// ---------------------------------------------------------------------------
// Kernel 2: We (E,D,O) fp32  ->  Wt (E,O,D) bf16  (transposed for B-frag loads)
// ---------------------------------------------------------------------------
__global__ __launch_bounds__(256) void moe_prep(const float* __restrict__ We,
                                                unsigned short* __restrict__ Wt) {
  int i = blockIdx.x * 256 + threadIdx.x;
  if (i < NE * ND * NO) {
    int e   = i / (ND * NO);
    int rem = i % (ND * NO);
    int d   = rem / NO;
    int o   = rem % NO;
    Wt[((size_t)e * NO + o) * ND + d] = f2bf(We[i]);
  }
}

// ---------------------------------------------------------------------------
// Kernel 2b (optional, if workspace is large enough):
// x (N,D) fp32 -> xb (N,D) bf16, so GEMM A-fragments are straight 16B loads.
// ---------------------------------------------------------------------------
__global__ __launch_bounds__(256) void moe_prep_x(const float* __restrict__ x,
                                                  unsigned short* __restrict__ xb) {
  size_t i = ((size_t)blockIdx.x * 256 + threadIdx.x) * 8;
  if (i < (size_t)NTOK * ND) {
    const float4* p = reinterpret_cast<const float4*>(x + i);
    float4 f0 = p[0], f1 = p[1];
    v8u r;
    r[0] = f2bf(f0.x); r[1] = f2bf(f0.y); r[2] = f2bf(f0.z); r[3] = f2bf(f0.w);
    r[4] = f2bf(f1.x); r[5] = f2bf(f1.y); r[6] = f2bf(f1.z); r[7] = f2bf(f1.w);
    *reinterpret_cast<v8u*>(xb + i) = r;
  }
}

// ---------------------------------------------------------------------------
// Kernel 3: gating -- one wave32 per token. Exact fp32 softmax + top-2,
// scatter (token, weight) into per-expert lists for the grouped GEMM.
// ---------------------------------------------------------------------------
__global__ __launch_bounds__(256) void moe_gate(const float* __restrict__ x,
                                                const float* __restrict__ Wg,
                                                const float* __restrict__ bg,
                                                int*   __restrict__ counts,
                                                int*   __restrict__ tlist,
                                                float* __restrict__ wlist) {
  int lane = threadIdx.x & 31;
  int t = (int)((blockIdx.x * (unsigned)blockDim.x + threadIdx.x) >> 5);
  if (t >= NTOK) return;
  const float* xr = x + (size_t)t * ND;

  float a0=0.f,a1=0.f,a2=0.f,a3=0.f,a4=0.f,a5=0.f,a6=0.f,a7=0.f;
  for (int d = lane; d < ND; d += 32) {
    float xv = xr[d];
    const float4* g4 = reinterpret_cast<const float4*>(Wg + d * NE);
    float4 g0 = g4[0], g1 = g4[1];
    a0 += xv * g0.x; a1 += xv * g0.y; a2 += xv * g0.z; a3 += xv * g0.w;
    a4 += xv * g1.x; a5 += xv * g1.y; a6 += xv * g1.z; a7 += xv * g1.w;
  }
#pragma unroll
  for (int off = 16; off >= 1; off >>= 1) {
    a0 += __shfl_xor(a0, off, 32); a1 += __shfl_xor(a1, off, 32);
    a2 += __shfl_xor(a2, off, 32); a3 += __shfl_xor(a3, off, 32);
    a4 += __shfl_xor(a4, off, 32); a5 += __shfl_xor(a5, off, 32);
    a6 += __shfl_xor(a6, off, 32); a7 += __shfl_xor(a7, off, 32);
  }
  if (lane == 0) {
    float p[NE] = {a0 + bg[0], a1 + bg[1], a2 + bg[2], a3 + bg[3],
                   a4 + bg[4], a5 + bg[5], a6 + bg[6], a7 + bg[7]};
    float mx = p[0];
#pragma unroll
    for (int e = 1; e < NE; e++) mx = fmaxf(mx, p[e]);
    float sum = 0.f;
#pragma unroll
    for (int e = 0; e < NE; e++) { p[e] = __expf(p[e] - mx); sum += p[e]; }
    float inv = 1.0f / sum;
    int i0 = 0;
#pragma unroll
    for (int e = 1; e < NE; e++) if (p[e] > p[i0]) i0 = e;      // lowest idx on tie
    int i1 = (i0 == 0) ? 1 : 0;
#pragma unroll
    for (int e = 0; e < NE; e++) if (e != i0 && p[e] > p[i1]) i1 = e;

    int pos0 = atomicAdd(counts + i0, 1);
    tlist[(size_t)i0 * NTOK + pos0] = t;
    wlist[(size_t)i0 * NTOK + pos0] = p[i0] * inv;
    int pos1 = atomicAdd(counts + i1, 1);
    tlist[(size_t)i1 * NTOK + pos1] = t;
    wlist[(size_t)i1 * NTOK + pos1] = p[i1] * inv;
  }
}

// ----- A-fragment helpers (contiguous k order used here, not the ISA table order; B matches):
// lanes 0-15 : K = kb + {0..15}
// lanes 16-31: K = kb + {16..31}
struct AFrag32 { float4 f0, f1, f2, f3; };
__device__ __forceinline__ AFrag32 loadA32(const float* xrow, int kb, int khalf) {
  AFrag32 r;
  const float4* p0 = reinterpret_cast<const float4*>(xrow + kb + khalf * 16);
  const float4* p1 = reinterpret_cast<const float4*>(xrow + kb + 8 + khalf * 16);
  r.f0 = p0[0]; r.f1 = p0[1]; r.f2 = p1[0]; r.f3 = p1[1];
  return r;
}
__device__ __forceinline__ v16bf cvtA32(const AFrag32& r) {
  v16u au;
  au[0]  = f2bf(r.f0.x); au[1]  = f2bf(r.f0.y); au[2]  = f2bf(r.f0.z); au[3]  = f2bf(r.f0.w);
  au[4]  = f2bf(r.f1.x); au[5]  = f2bf(r.f1.y); au[6]  = f2bf(r.f1.z); au[7]  = f2bf(r.f1.w);
  au[8]  = f2bf(r.f2.x); au[9]  = f2bf(r.f2.y); au[10] = f2bf(r.f2.z); au[11] = f2bf(r.f2.w);
  au[12] = f2bf(r.f3.x); au[13] = f2bf(r.f3.y); au[14] = f2bf(r.f3.z); au[15] = f2bf(r.f3.w);
  return __builtin_bit_cast(v16bf, au);
}
struct AFragB { v8u l0, l1; };
__device__ __forceinline__ AFragB loadAB(const unsigned short* xrow, int kb, int khalf) {
  AFragB r;
  r.l0 = *reinterpret_cast<const v8u*>(xrow + kb + khalf * 16);
  r.l1 = *reinterpret_cast<const v8u*>(xrow + kb + 8 + khalf * 16);
  return r;
}
__device__ __forceinline__ v16bf cvtAB(const AFragB& r) {
  v16u au = __builtin_shufflevector(r.l0, r.l1,
      0,1,2,3,4,5,6,7,8,9,10,11,12,13,14,15);
  return __builtin_bit_cast(v16bf, au);
}

// B layout: lane holds col n = lane&15; K = kb + (lane>=16)*16 + {0..15}
// contiguous -> one 32B load per fragment from Wt[e][n][k].
#define LOAD_B(DST, KB) do {                                                \
    _Pragma("unroll")                                                       \
    for (int s = 0; s < 8; s++)                                             \
      (DST)[s] = *reinterpret_cast<const v16u*>(                            \
          wbase + (size_t)s * (16 * ND) + (KB));                            \
  } while (0)

#define WMMA_ALL(AV, BB) do {                                               \
    v16bf _a = (AV);                                                        \
    _Pragma("unroll")                                                       \
    for (int s = 0; s < 8; s++)                                             \
      acc[s] = __builtin_amdgcn_wmma_f32_16x16x32_bf16(                     \
          false, _a, false, __builtin_bit_cast(v16bf, (BB)[s]),             \
          (short)0, acc[s], false, false);                                  \
  } while (0)

// ---------------------------------------------------------------------------
// Kernel 4: grouped GEMM per expert using v_wmma_f32_16x16x32_bf16.
// Block = 256 threads (8 waves). Tile = 64 gathered tokens x 256 outputs.
// Wave (wm = wave&3, wn = wave>>2) owns 16 tokens x 128 cols = 8 accum tiles.
// K-loop is software pipelined: double-buffered A/B fragments so the next
// k-step's 18 global loads overlap the current 8 WMMAs.
// ---------------------------------------------------------------------------
template<bool USE_XB>
__global__ __launch_bounds__(256) void moe_gemm(const float* __restrict__ x,
                                                const unsigned short* __restrict__ xb,
                                                const unsigned short* __restrict__ Wt,
                                                const float* __restrict__ be,
                                                const int*   __restrict__ counts,
                                                const int*   __restrict__ tlist,
                                                const float* __restrict__ wlist,
                                                float* __restrict__ out) {
  const int e    = blockIdx.y;
  const int cnt  = counts[e];
  const int row0 = blockIdx.x * MT;
  if (row0 >= cnt) return;

  __shared__ int   s_tok[MT];
  __shared__ float s_w[MT];
  const int tid = threadIdx.x;
  if (tid < MT) {
    int r = row0 + tid;
    if (r < cnt) {
      s_tok[tid] = tlist[(size_t)e * NTOK + r];
      s_w[tid]   = wlist[(size_t)e * NTOK + r];
    } else {
      s_tok[tid] = tlist[(size_t)e * NTOK + row0];  // safe clamp
      s_w[tid]   = 0.0f;                            // gated out at store
    }
  }
  __syncthreads();

  const int wave  = tid >> 5;
  const int lane  = tid & 31;
  const int wm    = wave & 3;        // M sub-tile (16 tokens)
  const int wn    = wave >> 2;       // N half (128 cols)
  const int m16   = lane & 15;
  const int khalf = lane >> 4;       // 0 or 1

  v8f acc[8];
#pragma unroll
  for (int s = 0; s < 8; s++) {
    v8f z = {0.f,0.f,0.f,0.f,0.f,0.f,0.f,0.f};
    acc[s] = z;
  }

  const int tokA = s_tok[wm * 16 + m16];
  const unsigned short* wbase =
      Wt + (size_t)e * NO * ND + (size_t)(wn * 128 + m16) * ND + khalf * 16;

  if constexpr (USE_XB) {
    const unsigned short* xrow = xb + (size_t)tokA * ND;
    AFragB a0 = loadAB(xrow, 0, khalf);
    v16u bb0[8]; LOAD_B(bb0, 0);
    for (int kb = 0; kb < ND; kb += 64) {
      AFragB a1 = loadAB(xrow, kb + 32, khalf);
      v16u bb1[8]; LOAD_B(bb1, kb + 32);
      WMMA_ALL(cvtAB(a0), bb0);
      if (kb + 64 < ND) { a0 = loadAB(xrow, kb + 64, khalf); LOAD_B(bb0, kb + 64); }
      WMMA_ALL(cvtAB(a1), bb1);
    }
  } else {
    const float* xrow = x + (size_t)tokA * ND;
    AFrag32 a0 = loadA32(xrow, 0, khalf);
    v16u bb0[8]; LOAD_B(bb0, 0);
    for (int kb = 0; kb < ND; kb += 64) {
      AFrag32 a1 = loadA32(xrow, kb + 32, khalf);
      v16u bb1[8]; LOAD_B(bb1, kb + 32);
      WMMA_ALL(cvtA32(a0), bb0);
      if (kb + 64 < ND) { a0 = loadA32(xrow, kb + 64, khalf); LOAD_B(bb0, kb + 64); }
      WMMA_ALL(cvtA32(a1), bb1);
    }
  }

  // ---- epilogue. C/D layout -> VGPR r holds row M = khalf*8 + r, col lane&15.
  // Preload this wave's 8 rows of (weight, token) once.
  float  wrow[8];
  size_t obase[8];
#pragma unroll
  for (int r = 0; r < 8; r++) {
    int mrow = wm * 16 + khalf * 8 + r;
    wrow[r]  = s_w[mrow];
    obase[r] = (size_t)s_tok[mrow] * NO;
  }
  // out[token] += gate_w * (gemm + bias); exactly 2 contributions per token
  // arrive from different blocks -> f32 atomic add into zeroed out.
#pragma unroll
  for (int s = 0; s < 8; s++) {
    int ncol = wn * 128 + s * 16 + m16;
    float bias = be[e * NO + ncol];
#pragma unroll
    for (int r = 0; r < 8; r++) {
      if (wrow[r] != 0.0f) {
        atomicAdd(out + obase[r] + ncol, wrow[r] * (acc[s][r] + bias));
      }
    }
  }
}

// ---------------------------------------------------------------------------
extern "C" void kernel_launch(void* const* d_in, const int* in_sizes, int n_in,
                              void* d_out, int out_size, void* d_ws, size_t ws_size,
                              hipStream_t stream) {
  const float* x  = (const float*)d_in[0];   // (8,4096,768)
  const float* Wg = (const float*)d_in[1];   // (768,8)
  const float* bg = (const float*)d_in[2];   // (8,)
  const float* We = (const float*)d_in[3];   // (8,768,256)
  const float* be = (const float*)d_in[4];   // (8,256)
  float* out = (float*)d_out;                // (8,4096,256)

  char* ws = (char*)d_ws;
  const size_t off_tlist = 256;
  const size_t off_wlist = off_tlist + (size_t)NE * NTOK * 4;   // +1 MB
  const size_t off_Wt    = off_wlist + (size_t)NE * NTOK * 4;   // +1 MB
  const size_t off_xb    = off_Wt + (size_t)NE * NO * ND * 2;   // +3 MB
  const size_t need_xb   = off_xb + (size_t)NTOK * ND * 2;      // +50.3 MB

  int*            counts = (int*)ws;
  int*            tlist  = (int*)(ws + off_tlist);
  float*          wlist  = (float*)(ws + off_wlist);
  unsigned short* Wt     = (unsigned short*)(ws + off_Wt);
  unsigned short* xb     = (unsigned short*)(ws + off_xb);
  const bool use_xb = (ws_size >= need_xb);   // ws_size constant -> deterministic

  // 1) zero out + counters
  moe_init<<<(NTOK * NO / 4 + 255) / 256, 256, 0, stream>>>(out, counts);
  // 2) pack weights to bf16 [e][o][d]
  moe_prep<<<(NE * ND * NO + 255) / 256, 256, 0, stream>>>(We, Wt);
  // 2b) optionally pack activations to bf16
  if (use_xb)
    moe_prep_x<<<(NTOK * ND / 8 + 255) / 256, 256, 0, stream>>>(x, xb);
  // 3) gate + routing (one wave per token, 8 tokens per block)
  moe_gate<<<NTOK / 8, 256, 0, stream>>>(x, Wg, bg, counts, tlist, wlist);
  // 4) grouped GEMM over routed tokens
  dim3 grid((NTOK + MT - 1) / MT, NE);
  if (use_xb)
    moe_gemm<true ><<<grid, 256, 0, stream>>>(x, xb, Wt, be, counts, tlist, wlist, out);
  else
    moe_gemm<false><<<grid, 256, 0, stream>>>(x, xb, Wt, be, counts, tlist, wlist, out);
}
